// NVFP4Linear_16655883173974
// MI455X (gfx1250) — compile-verified, matrix operand order checked by reading, not run on hardware
//
#include <hip/hip_runtime.h>

typedef __attribute__((ext_vector_type(16))) __bf16 v16bf;
typedef __attribute__((ext_vector_type(8)))  __bf16 v8bf;
typedef __attribute__((ext_vector_type(2)))  __bf16 v2bf;
typedef __attribute__((ext_vector_type(8)))  float  v8f;

#define M_DIM     64
#define N_DIM     8192
#define K_DIM     8192
#define KP_ROW    (K_DIM / 2)        // int32 per weight row (2 fp4 per int32)
#define SB_ROW    (K_DIM / 16)       // scale blocks per row
#define SPLITK    8
#define KCHUNK    (K_DIM / SPLITK)   // 1024
#define KSTEP     32                 // WMMA K for bf16
#define KPANEL    256                // staged x panel width (8 WMMA steps)
#define XROW      (KPANEL + 8)       // +8 halfs pad: 528B row -> 4-bank skew

// E2M1 nibble -> signed f32 (used once per block to build the LDS table)
__device__ __forceinline__ float fp4_val(unsigned c) {
    unsigned m  = c & 7u;
    unsigned em = ((126u + (m >> 1)) << 23) | ((m & 1u) << 22);
    unsigned bits = (m >= 2u) ? em : (m == 1u ? 0x3F000000u : 0u);
    bits |= (c & 8u) << 28;
    return __uint_as_float(bits);
}

__device__ __forceinline__ __bf16 to_bf16(float f) { return (__bf16)f; }

__global__ void __launch_bounds__(256)
nvfp4_init_out(const float* __restrict__ bias, float* __restrict__ out) {
    int i = blockIdx.x * blockDim.x + threadIdx.x;   // 64*8192 threads
    out[i] = bias[i & (N_DIM - 1)];
}

__global__ void __launch_bounds__(256)
nvfp4_wmma_kernel(const float* __restrict__ x,
                  const int*   __restrict__ wp,
                  const float* __restrict__ wsc,
                  const float* __restrict__ gsc,
                  float*       __restrict__ out) {
    __shared__ __align__(16) __bf16 xs[M_DIM * XROW];  // 33 KB staged x panel
    __shared__ v2bf tbl[256];                          // byte -> 2 signed bf16

    const int tid    = threadIdx.x;
    const int lane   = tid & 31;
    const int wave   = tid >> 5;                       // 0..7
    const int laneHi = lane >> 4;
    const int laneLo = lane & 15;

    // ---- build dequant table: byte b -> {val(b&15), val(b>>4)} (branchy, once)
    {
        v2bf e;
        e[0] = to_bf16(fp4_val((unsigned)tid & 15u));
        e[1] = to_bf16(fp4_val(((unsigned)tid >> 4) & 15u));
        tbl[tid] = e;
    }

    const int nb  = blockIdx.x & 63;                   // N-tile group
    const int ks  = blockIdx.x >> 6;                   // K split id (0..7)
    const int n0  = nb * 128 + wave * 16;
    const int kc0 = ks * KCHUNK;

    const float g    = gsc[0];
    const float invg = (g != 0.0f) ? (1.0f / g) : 1.0f;

    const int    n    = n0 + laneLo;                   // weight row = out col
    const int*   wrow = wp  + (size_t)n * KP_ROW;
    const float* srow = wsc + (size_t)n * SB_ROW;

    // x staging: thread -> 64 contiguous f32 of the 64 x 256 panel
    const int tm = tid >> 2;                           // M row 0..63
    const int tk = (tid & 3) * 64;                     // 0,64,128,192

    v8f acc[4];
    #pragma unroll
    for (int t = 0; t < 4; ++t) acc[t] = (v8f)0.0f;

    const int koff = laneHi ? 8 : 0;

    for (int p = 0; p < KCHUNK / KPANEL; ++p) {
        const int kp = kc0 + p * KPANEL;

        __syncthreads();   // prior panel reads (and table writes) complete
        #pragma unroll
        for (int j = 0; j < 8; ++j) {                  // 8 f32 per store
            const float4* xg = (const float4*)(x + (size_t)tm * K_DIM + kp + tk + j * 8);
            float4 xa = xg[0];
            float4 xb = xg[1];
            v8bf xh;
            xh[0] = to_bf16(xa.x); xh[1] = to_bf16(xa.y);
            xh[2] = to_bf16(xa.z); xh[3] = to_bf16(xa.w);
            xh[4] = to_bf16(xb.x); xh[5] = to_bf16(xb.y);
            xh[6] = to_bf16(xb.z); xh[7] = to_bf16(xb.w);
            *(v8bf*)(xs + tm * XROW + tk + j * 8) = xh; // ds_store_b128
        }
        __syncthreads();

        #pragma unroll   // full unroll: lets next step's global loads hoist
        for (int s = 0; s < KPANEL; s += KSTEP) {      // behind this step's WMMAs
            const int k = kp + s;

            // A fragments first, distinct regs: in-order DS countdown for the
            // table lookups below retires these too (no per-WMMA dscnt drain)
            v16bf a[4];
            #pragma unroll
            for (int mt = 0; mt < 4; ++mt) {
                const __bf16* xr = xs + (mt * 16 + laneLo) * XROW + s + koff;
                v8bf lo = *(const v8bf*)(xr);          // K = koff..koff+7
                v8bf hi = *(const v8bf*)(xr + 16);     // K = 16+koff..
                #pragma unroll
                for (int e = 0; e < 8; ++e) { a[mt][e] = lo[e]; a[mt][8 + e] = hi[e]; }
            }

            // weights: 2 x b128 covers this lane's 16 nibbles of the 32-K slab
            const int  i0 = (k >> 1) + (laneHi << 2);
            const int4 w0 = *(const int4*)(wrow + i0);
            const int4 w1 = *(const int4*)(wrow + i0 + 8);
            __builtin_prefetch(wrow + i0 + 128, 0, 3); // next panel's stream

            // block scales for the two 16-wide K blocks
            const float2 sc = *(const float2*)(srow + (k >> 4));
            const __bf16 sa = to_bf16(sc.x * invg);
            const __bf16 sb = to_bf16(sc.y * invg);
            const v2bf spA = {sa, sa};
            const v2bf spB = {sb, sb};

            // dequant via table: ds_load_b32 + v_pk_mul_bf16 per int32
            v16bf b;
            v2bf* bp = (v2bf*)&b;
            #pragma unroll
            for (int j = 0; j < 4; ++j) {
                unsigned w = ((const unsigned*)&w0)[j];
                bp[j] = tbl[w & 255u] * spA;
            }
            #pragma unroll
            for (int j = 0; j < 4; ++j) {
                unsigned w = ((const unsigned*)&w1)[j];
                bp[4 + j] = tbl[w & 255u] * spB;
            }

            // 4 back-to-back WMMAs (A regs all resident)
            #pragma unroll
            for (int mt = 0; mt < 4; ++mt) {
                acc[mt] = __builtin_amdgcn_wmma_f32_16x16x32_bf16(
                    false, a[mt], false, b, (short)0, acc[mt], false, false);
            }
        }
    }

    // combine split-K partials onto bias-initialized out
    #pragma unroll
    for (int mt = 0; mt < 4; ++mt) {
        #pragma unroll
        for (int r = 0; r < 8; ++r) {
            const int mrow = mt * 16 + r + (laneHi ? 8 : 0);
            unsafeAtomicAdd(out + (size_t)mrow * N_DIM + n0 + laneLo, acc[mt][r]);
        }
    }
}

extern "C" void kernel_launch(void* const* d_in, const int* in_sizes, int n_in,
                              void* d_out, int out_size, void* d_ws, size_t ws_size,
                              hipStream_t stream) {
    const float* x    = (const float*)d_in[0];
    const int*   wp   = (const int*)  d_in[1];
    const float* wsc  = (const float*)d_in[2];
    const float* gsc  = (const float*)d_in[3];
    const float* bias = (const float*)d_in[4];
    float* out = (float*)d_out;

    nvfp4_init_out<<<(M_DIM * N_DIM) / 256, 256, 0, stream>>>(bias, out);
    nvfp4_wmma_kernel<<<64 * SPLITK, 256, 0, stream>>>(x, wp, wsc, gsc, out);
}
